// Res_Graph_Conv_Lyr_6545530159681
// MI455X (gfx1250) — matrix operand order repaired and checked by reading, not compiled, not run on hardware
//
#include <hip/hip_runtime.h>

#define N_NODES 50000
#define N_EDGES 400000
#define IN_C 32
#define OUT_C 32
#define EDGE_DIM 16
#define HIDDEN 128
#define BN_EPS 1e-5f
#define NW 8                          // waves per block in edge kernel
#define TGROUP 4                      // edge tiles (of 16) processed together per wave
#define NTILES (N_EDGES / 16)         // 25000
#define NGROUPS (NTILES / TGROUP)     // 6250

typedef _Float16 h16;
typedef __attribute__((ext_vector_type(16))) _Float16 v16h;
typedef __attribute__((ext_vector_type(8)))  _Float16 v8h;
typedef __attribute__((ext_vector_type(8)))  float    v8f;
typedef __attribute__((ext_vector_type(4)))  float    f32x4;

static __device__ __forceinline__ v16h concat8(v8h lo, v8h hi) {
    union { struct { v8h a, b; } s; v16h v; } u;
    u.s.a = lo; u.s.b = hi;
    return u.v;
}

static __device__ __forceinline__ void atomAddF(float* p, float v) {
    __hip_atomic_fetch_add(p, v, __ATOMIC_RELAXED, __HIP_MEMORY_SCOPE_AGENT);
}

// ---------------- kernel 0: zero workspace accumulators ----------------
__global__ void k_zero(float* __restrict__ agg, float* __restrict__ cnt,
                       float* __restrict__ stats) {
    int i = blockIdx.x * blockDim.x + threadIdx.x;
    int stride = gridDim.x * blockDim.x;
    for (int j = i; j < N_NODES * OUT_C; j += stride) agg[j] = 0.0f;
    for (int j = i; j < N_NODES; j += stride) cnt[j] = 0.0f;
    if (i < 2 * OUT_C) stats[i] = 0.0f;
}

// ------- kernel 1: transpose+convert W2 [128,1024]f32 -> [1024,128]f16 (L2-resident) -------
__global__ void k_w2t(const float* __restrict__ w2, h16* __restrict__ w2t) {
    int i = blockIdx.x * blockDim.x + threadIdx.x;
    int stride = gridDim.x * blockDim.x;
    for (int idx = i; idx < HIDDEN * IN_C * OUT_C; idx += stride) {
        int k = idx >> 10;          // row in W2 (0..127)
        int col = idx & 1023;       // column = i*32 + o
        w2t[(size_t)col * HIDDEN + k] = (h16)w2[idx];
    }
}

// ---------------- kernel 2: fused edge MLP + NNConv message + scatter ----------------
__global__ __launch_bounds__(32 * NW) void k_edge(
    const float* __restrict__ x, const int* __restrict__ eidx,
    const float* __restrict__ eattr,
    const float* __restrict__ w1, const float* __restrict__ b1,
    const float* __restrict__ b2, const h16* __restrict__ w2t,
    float* __restrict__ agg, float* __restrict__ cnt) {

    extern __shared__ char smem[];
    float* W1s = (float*)smem;                       // 16*128 f32
    float* b1s = W1s + EDGE_DIM * HIDDEN;            // 128 f32
    float* b2s = b1s + HIDDEN;                       // 1024 f32
    h16*   Hs  = (h16*)(b2s + IN_C * OUT_C);         // NW * TGROUP * 16 * 128 f16

    for (int j = threadIdx.x; j < EDGE_DIM * HIDDEN; j += blockDim.x) W1s[j] = w1[j];
    for (int j = threadIdx.x; j < HIDDEN; j += blockDim.x) b1s[j] = b1[j];
    for (int j = threadIdx.x; j < IN_C * OUT_C; j += blockDim.x) b2s[j] = b2[j];
    __syncthreads();

    const int wave = threadIdx.x >> 5;
    const int lane = threadIdx.x & 31;
    const int half = lane >> 4;       // 0 or 1
    const int sub  = lane & 15;       // row/col index within 16
    h16* Hw = Hs + (size_t)wave * TGROUP * 16 * HIDDEN;

    for (int g = blockIdx.x * NW + wave; g < NGROUPS; g += gridDim.x * NW) {
        int srcs[TGROUP], dsts[TGROUP];

        // ---- stage 1 (VALU, ~1.5% of FLOPs): H[e,k] = relu(ea @ W1 + b1) -> LDS f16
        #pragma unroll
        for (int q = 0; q < TGROUP; ++q) {
            const int eg = (g * TGROUP + q) * 16 + sub;
            srcs[q] = eidx[eg];
            dsts[q] = eidx[N_EDGES + eg];

            float ea[EDGE_DIM];
            {
                const f32x4* p = (const f32x4*)(eattr + (size_t)eg * EDGE_DIM);
                #pragma unroll
                for (int qq = 0; qq < 4; ++qq) {
                    f32x4 v = p[qq];
                    #pragma unroll
                    for (int r = 0; r < 4; ++r) ea[4 * qq + r] = v[r];
                }
            }
            const int k0 = half * 64;   // this half computes k in [k0, k0+64)
            #pragma unroll
            for (int kb = 0; kb < 64; kb += 8) {
                float acc[8];
                {
                    const f32x4* bp = (const f32x4*)&b1s[k0 + kb];
                    f32x4 a0 = bp[0], a1 = bp[1];
                    #pragma unroll
                    for (int r = 0; r < 4; ++r) { acc[r] = a0[r]; acc[4 + r] = a1[r]; }
                }
                #pragma unroll
                for (int j = 0; j < EDGE_DIM; ++j) {
                    const f32x4* wp = (const f32x4*)&W1s[j * HIDDEN + k0 + kb];
                    f32x4 wa = wp[0], wb = wp[1];
                    float e = ea[j];
                    #pragma unroll
                    for (int r = 0; r < 4; ++r) {
                        acc[r]     = fmaf(e, wa[r], acc[r]);
                        acc[4 + r] = fmaf(e, wb[r], acc[4 + r]);
                    }
                }
                v8h hv;
                #pragma unroll
                for (int r = 0; r < 8; ++r) hv[r] = (h16)fmaxf(acc[r], 0.0f);
                *(v8h*)&Hw[q * 16 * HIDDEN + sub * HIDDEN + k0 + kb] = hv;
            }
        }
        __threadfence_block();   // drain DScnt: lanes read H written by partner lanes

        // ---- stage 2: msg[o,e] over input channels i; each A fragment feeds TGROUP WMMAs
        v8f acc[TGROUP][2];
        #pragma unroll
        for (int q = 0; q < TGROUP; ++q)
            #pragma unroll
            for (int t = 0; t < 2; ++t)
                acc[q][t] = (v8f){0, 0, 0, 0, 0, 0, 0, 0};

        const float* xr[TGROUP];
        #pragma unroll
        for (int q = 0; q < TGROUP; ++q) xr[q] = x + (size_t)srcs[q] * IN_C;

        for (int i = 0; i < IN_C; ++i) {
            float xv[TGROUP];
            #pragma unroll
            for (int q = 0; q < TGROUP; ++q) xv[q] = xr[q][i];

            const h16* wcolA = w2t + ((size_t)i * 32) * HIDDEN;
            #pragma unroll
            for (int t = 0; t < 2; ++t) {
                // C seeded with b2 (rows o = t*16 + half*8 + r; same for every lane/tile)
                v8f ci;
                {
                    const f32x4* bp = (const f32x4*)&b2s[i * 32 + t * 16 + half * 8];
                    f32x4 c0 = bp[0], c1 = bp[1];
                    #pragma unroll
                    for (int r = 0; r < 4; ++r) { ci[r] = c0[r]; ci[4 + r] = c1[r]; }
                }
                v8f c[TGROUP];
                #pragma unroll
                for (int q = 0; q < TGROUP; ++q) c[q] = ci;

                // chain K=128 as 4 x (16x16x32); A = W2^T tile (row o, K k), shared by 4 tiles
                #pragma unroll
                for (int cc = 0; cc < 4; ++cc) {
                    const h16* ap = wcolA + ((size_t)(t * 16 + sub)) * HIDDEN
                                    + cc * 32 + half * 8;
                    v8h alo = *(const v8h*)ap;          // elems 0..7 : K = 8*half + j
                    v8h ahi = *(const v8h*)(ap + 16);   // elems 8..15: K = 16 + 8*half + j
                    v16h a = concat8(alo, ahi);
                    #pragma unroll
                    for (int q = 0; q < TGROUP; ++q) {
                        // B: H as 32x16 (K x edge), lane = edge column, K = 8*half + j / 16 + 8*half + j
                        const h16* bp = &Hw[q * 16 * HIDDEN + sub * HIDDEN
                                                    + cc * 32 + half * 8];
                        v16h bq = concat8(*(const v8h*)bp, *(const v8h*)(bp + 16));
                        c[q] = __builtin_amdgcn_wmma_f32_16x16x32_f16(
                                   false, a, false, bq, (short)0, c[q], false, false);
                    }
                }
                #pragma unroll
                for (int q = 0; q < TGROUP; ++q) acc[q][t] += xv[q] * c[q];
            }
        }

        // ---- scatter-add messages (lane holds edge column e=sub, rows o = t*16+half*8+r)
        #pragma unroll
        for (int q = 0; q < TGROUP; ++q) {
            float* arow = agg + (size_t)dsts[q] * OUT_C;
            #pragma unroll
            for (int r = 0; r < 8; ++r) {
                atomAddF(&arow[half * 8 + r],      acc[q][0][r]);
                atomAddF(&arow[16 + half * 8 + r], acc[q][1][r]);
            }
            if (half == 0) atomAddF(&cnt[dsts[q]], 1.0f);
        }
    }
}

// ---------------- kernel 3: h = agg/max(cnt,1) + x@root_w + bias; BN stats ----------------
__global__ __launch_bounds__(256) void k_node(
    const float* __restrict__ x, const float* __restrict__ rootw,
    const float* __restrict__ cbias,
    float* __restrict__ agg /* in: sums, out: h */,
    const float* __restrict__ cnt, float* __restrict__ stats) {

    __shared__ float rws[IN_C * OUT_C];
    __shared__ float bs[OUT_C];
    for (int j = threadIdx.x; j < IN_C * OUT_C; j += blockDim.x) rws[j] = rootw[j];
    if (threadIdx.x < OUT_C) bs[threadIdx.x] = cbias[threadIdx.x];
    __syncthreads();

    int n = blockIdx.x * blockDim.x + threadIdx.x;
    bool act = n < N_NODES;
    float h[OUT_C];
    #pragma unroll
    for (int o = 0; o < OUT_C; ++o) h[o] = 0.0f;

    if (act) {
        float inv = 1.0f / fmaxf(cnt[n], 1.0f);
        const float* ar = agg + (size_t)n * OUT_C;
        const float* xr = x + (size_t)n * IN_C;
        #pragma unroll
        for (int o = 0; o < OUT_C; ++o) h[o] = ar[o] * inv + bs[o];
        #pragma unroll 4
        for (int i = 0; i < IN_C; ++i) {
            float xv = xr[i];
            #pragma unroll
            for (int o = 0; o < OUT_C; ++o) h[o] = fmaf(xv, rws[i * OUT_C + o], h[o]);
        }
        float* aw = agg + (size_t)n * OUT_C;
        #pragma unroll
        for (int o = 0; o < OUT_C; ++o) aw[o] = h[o];
    }

    // wave32 reduction of sum(h) and sum(h^2) per channel, then global atomics
    #pragma unroll
    for (int o = 0; o < OUT_C; ++o) {
        float s1 = act ? h[o] : 0.0f;
        float s2 = act ? h[o] * h[o] : 0.0f;
        #pragma unroll
        for (int off = 16; off > 0; off >>= 1) {
            s1 += __shfl_down(s1, off, 32);
            s2 += __shfl_down(s2, off, 32);
        }
        if ((threadIdx.x & 31) == 0) {
            atomAddF(&stats[o], s1);
            atomAddF(&stats[OUT_C + o], s2);
        }
    }
}

// ---------------- kernel 4: batchnorm + relu + residual ----------------
__global__ __launch_bounds__(256) void k_final(
    const float* __restrict__ x, const float* __restrict__ h,
    const float* __restrict__ stats,
    const float* __restrict__ gamma, const float* __restrict__ beta,
    float* __restrict__ out) {

    __shared__ float mu[OUT_C], rs[OUT_C], gs[OUT_C], bs[OUT_C];
    if (threadIdx.x < OUT_C) {
        float m = stats[threadIdx.x] * (1.0f / N_NODES);
        float v = stats[OUT_C + threadIdx.x] * (1.0f / N_NODES) - m * m;
        mu[threadIdx.x] = m;
        rs[threadIdx.x] = rsqrtf(v + BN_EPS);
        gs[threadIdx.x] = gamma[threadIdx.x];
        bs[threadIdx.x] = beta[threadIdx.x];
    }
    __syncthreads();

    int n = blockIdx.x * blockDim.x + threadIdx.x;
    if (n < N_NODES) {
        const float* hr = h + (size_t)n * OUT_C;
        const float* xr = x + (size_t)n * IN_C;
        float* orow = out + (size_t)n * OUT_C;
        #pragma unroll
        for (int o = 0; o < OUT_C; ++o) {
            float hn = (hr[o] - mu[o]) * rs[o] * gs[o] + bs[o];
            orow[o] = xr[o] + fmaxf(hn, 0.0f);
        }
    }
}

extern "C" void kernel_launch(void* const* d_in, const int* in_sizes, int n_in,
                              void* d_out, int out_size, void* d_ws, size_t ws_size,
                              hipStream_t stream) {
    const float* x     = (const float*)d_in[0];
    const int*   eidx  = (const int*)d_in[1];    // edge_index [2, E]
    const float* eattr = (const float*)d_in[2];
    const float* w1    = (const float*)d_in[3];
    const float* b1    = (const float*)d_in[4];
    const float* w2    = (const float*)d_in[5];
    const float* b2    = (const float*)d_in[6];
    const float* rootw = (const float*)d_in[7];
    const float* cbias = (const float*)d_in[8];
    const float* gamma = (const float*)d_in[9];
    const float* beta  = (const float*)d_in[10];
    float* out = (float*)d_out;

    char* ws = (char*)d_ws;
    h16*   w2t   = (h16*)ws;                                           // 256 KB
    float* agg   = (float*)(ws + (size_t)IN_C * OUT_C * HIDDEN * sizeof(h16));
    float* cnt   = agg + (size_t)N_NODES * OUT_C;
    float* stats = cnt + N_NODES;

    k_zero<<<512, 256, 0, stream>>>(agg, cnt, stats);
    k_w2t<<<256, 256, 0, stream>>>(w2, w2t);

    size_t smem = (size_t)(EDGE_DIM * HIDDEN + HIDDEN + IN_C * OUT_C) * sizeof(float)
                + (size_t)NW * TGROUP * 16 * HIDDEN * sizeof(h16);
    k_edge<<<200, 32 * NW, smem, stream>>>(x, eidx, eattr, w1, b1, b2, w2t, agg, cnt);

    int nb = (N_NODES + 255) / 256;
    k_node<<<nb, 256, 0, stream>>>(x, rootw, cbias, agg, cnt, stats);
    k_final<<<nb, 256, 0, stream>>>(x, agg, stats, gamma, beta, out);
}
